// Attention_19653770346884
// MI455X (gfx1250) — matrix operand order repaired and checked by reading, not compiled, not run on hardware
//
#include <hip/hip_runtime.h>

#define Bb 2
#define Ss 2048
#define Ee 1024
#define Hh 16
#define Dd 64

typedef __attribute__((ext_vector_type(16))) _Float16 v16h;
typedef __attribute__((ext_vector_type(8)))  float    v8f;

// One WMMA operand fragment: 16 halves per lane = two 16-byte chunks.
union AF { v16h v; uint4 u[2]; };

__device__ __forceinline__ v8f wmma16(const AF& a, const AF& b, v8f c) {
  // D = A(16x32 f16) * B(32x16 f16) + C(16x16 f32)
  return __builtin_amdgcn_wmma_f32_16x16x32_f16(false, a.v, false, b.v,
                                                (short)0, c, false, false);
}

// CDNA5 async global->LDS copy (ASYNCcnt-tracked, bypasses VGPRs).
__device__ __forceinline__ void async_ld_b128(void* lds, const void* g) {
  unsigned l = (unsigned)(uintptr_t)(__attribute__((address_space(3))) void*)lds;
  unsigned long long ga = (unsigned long long)(uintptr_t)g;
  asm volatile("global_load_async_to_lds_b128 %0, %1, off"
               :: "v"(l), "v"(ga) : "memory");
}
__device__ __forceinline__ void wait_async0() {
  asm volatile("s_wait_asynccnt 0" ::: "memory");
}
__device__ __forceinline__ void wait_async4() {
  asm volatile("s_wait_asynccnt 4" ::: "memory");
}

// ---------------------------------------------------------------------------
// Kernel 1: fused QKV projection.  C[4096 x 1024] = x[4096 x 1024] * W + b
// grid = (32, 8 head-pairs, 3 {q,k,v}); block = 256 (8 waves).
// Tile 128x128 (2 heads), wave tile 32x64 -> 8 WMMAs per K-step per wave.
// Ping-pong LDS + register-staged loads: next tile's global loads are issued
// before the current tile's WMMAs so HBM latency hides behind the matrix pipe.
// Q,K stored f16 [B,H,S,D]; V stored f16 transposed [B,H,D,S].
// ---------------------------------------------------------------------------
__global__ __launch_bounds__(256, 2) void k_qkv(
    const float* __restrict__ x,
    const float* __restrict__ Wq, const float* __restrict__ Wk,
    const float* __restrict__ Wv,
    const float* __restrict__ bq, const float* __restrict__ bk,
    const float* __restrict__ bv,
    _Float16* __restrict__ Qh, _Float16* __restrict__ Kh,
    _Float16* __restrict__ Vh)
{
  __shared__ __align__(16) _Float16 As[2][128][32];   // [m][k] row-major
  __shared__ __align__(16) _Float16 Bs[2][128][32];   // [n][k] (B transposed)

  const int kind = blockIdx.z;
  const float* __restrict__ W  = (kind == 0) ? Wq : (kind == 1 ? Wk : Wv);
  const float* __restrict__ bi = (kind == 0) ? bq : (kind == 1 ? bk : bv);
  _Float16* __restrict__ Out   = (kind == 0) ? Qh : (kind == 1 ? Kh : Vh);

  const int h0   = blockIdx.y * 2;      // two heads per block (BN = 128)
  const int bm0  = blockIdx.x * 128;
  const int tid  = threadIdx.x;
  const int wave = tid >> 5;
  const int lane = tid & 31;
  const int half = lane >> 4;
  const int ln   = lane & 15;
  const int wm   = wave & 3;            // 4 M-groups of 32 rows
  const int wn   = wave >> 2;           // 2 N-groups of 64 cols
  const int kbase = half * 8;           // f16 A/B fragment K-interleave

  float4 fa[4], fb[4];                  // register-staged tiles
  auto loadRegs = [&](int k0) {
#pragma unroll
    for (int i = 0; i < 4; ++i) {       // A: 128x32 f32
      int idx = tid + i * 256, row = idx >> 3, c4 = idx & 7;
      fa[i] = *(const float4*)(x + (size_t)(bm0 + row) * Ee + k0 + c4 * 4);
    }
#pragma unroll
    for (int i = 0; i < 4; ++i) {       // B: 32(e) x 128(n) across 2 heads
      int idx = tid + i * 256, e = idx >> 5, c = idx & 31;
      int h = h0 + (c >> 4), d = (c & 15) * 4;
      fb[i] = *(const float4*)(W + ((size_t)h * Ee + k0 + e) * Dd + d);
    }
  };
  auto storeRegs = [&](int p) {
#pragma unroll
    for (int i = 0; i < 4; ++i) {
      int idx = tid + i * 256, row = idx >> 3, c4 = idx & 7;
      _Float16* d = &As[p][row][c4 * 4];
      d[0] = (_Float16)fa[i].x; d[1] = (_Float16)fa[i].y;
      d[2] = (_Float16)fa[i].z; d[3] = (_Float16)fa[i].w;
    }
#pragma unroll
    for (int i = 0; i < 4; ++i) {        // transpose: Bs[n][k=e]
      int idx = tid + i * 256, e = idx >> 5, c = idx & 31;
      Bs[p][c * 4 + 0][e] = (_Float16)fb[i].x;
      Bs[p][c * 4 + 1][e] = (_Float16)fb[i].y;
      Bs[p][c * 4 + 2][e] = (_Float16)fb[i].z;
      Bs[p][c * 4 + 3][e] = (_Float16)fb[i].w;
    }
  };

  loadRegs(0);
  storeRegs(0);
  __syncthreads();

  v8f acc[2][4] = {};
  for (int k0 = 0; k0 < Ee; k0 += 32) {
    const int p   = (k0 >> 5) & 1;
    const bool nxt = (k0 + 32) < Ee;
    if (nxt) loadRegs(k0 + 32);          // overlap global loads with WMMA
    if (k0 + 64 < Ee)                    // gfx1250 global_prefetch_b8
      __builtin_prefetch(x + (size_t)(bm0 + (tid >> 3)) * Ee + k0 + 64, 0, 1);

    AF a[2], bf[4];
#pragma unroll
    for (int t = 0; t < 2; ++t) {
      const int row = wm * 32 + t * 16 + ln;
      a[t].u[0] = *(const uint4*)&As[p][row][kbase];
      a[t].u[1] = *(const uint4*)&As[p][row][16 + kbase];
    }
#pragma unroll
    for (int t = 0; t < 4; ++t) {
      const int col = wn * 64 + t * 16 + ln;
      bf[t].u[0] = *(const uint4*)&Bs[p][col][kbase];
      bf[t].u[1] = *(const uint4*)&Bs[p][col][16 + kbase];
    }
#pragma unroll
    for (int tm = 0; tm < 2; ++tm)
#pragma unroll
      for (int tn = 0; tn < 4; ++tn)
        acc[tm][tn] = wmma16(a[tm], bf[tn], acc[tm][tn]);

    if (nxt) {
      storeRegs(p ^ 1);                  // convert + ds_store into other buffer
      __syncthreads();
    }
  }

  // Epilogue: +bias, f32 -> f16 store (C-layout: lane=col, VGPR r=row).
#pragma unroll
  for (int tm = 0; tm < 2; ++tm) {
#pragma unroll
    for (int tn = 0; tn < 4; ++tn) {
      const int n = wn * 64 + tn * 16 + ln;       // 0..127
      const int h = h0 + (n >> 6);
      const int d = n & 63;
      const float bias = bi[h * Dd + d];
#pragma unroll
      for (int r = 0; r < 8; ++r) {
        const int mg = bm0 + wm * 32 + tm * 16 + r + 8 * half;
        const int bb = mg >> 11;                  // / 2048
        const int s  = mg & 2047;
        const float v = acc[tm][tn][r] + bias;
        if (kind == 2)   // V transposed [B,H,D,S] for flash B-fragments
          Out[(((size_t)bb * Hh + h) * Dd + d) * Ss + s] = (_Float16)v;
        else             // Q,K row-major [B,H,S,D]
          Out[(((size_t)bb * Hh + h) * Ss + s) * Dd + d] = (_Float16)v;
      }
    }
  }
}

// ---------------------------------------------------------------------------
// Kernel 2: causal flash attention per (b,h,64-row query tile).
// block = 128 (4 waves); wave owns 16 query rows; keys streamed 32 at a time.
// K/V tiles are pure f16 copies -> CDNA5 async global->LDS, double-buffered.
// All WMMA-guarding conditions are readfirstlane-scalarized so the compiler
// lowers them as SGPR branches and EXEC stays all-ones around WMMA.
// ---------------------------------------------------------------------------
__global__ __launch_bounds__(128, 2) void k_flash(
    const _Float16* __restrict__ Qh, const _Float16* __restrict__ Kh,
    const _Float16* __restrict__ Vh, _Float16* __restrict__ attn)
{
  __shared__ __align__(16) _Float16 Ks[2][32][64];   // K block, [t][d]
  __shared__ __align__(16) _Float16 Vt[2][64][32];   // V block transposed [d][t]
  __shared__ __align__(16) _Float16 Ps[4][16][32];   // per-wave P (A-layout src)

  const int m0 = blockIdx.x * 64;
  const int h  = blockIdx.y;
  const int b  = blockIdx.z;

  const int tid  = threadIdx.x;
  const int lane = tid & 31;
  const int half = lane >> 4, ln = lane & 15;
  const int kbase = half * 8;
  // Scalarized wave id: guards below become SGPR compares (EXEC untouched).
  const int wave = __builtin_amdgcn_readfirstlane(tid >> 5);
  const int wrow0 = m0 + wave * 16;
  const int wrow_max = wrow0 + 15;

  const _Float16* Qb = Qh + ((size_t)b * Hh + h) * Ss * Dd;
  const _Float16* Kb = Kh + ((size_t)b * Hh + h) * Ss * Dd;
  const _Float16* Vb = Vh + ((size_t)b * Hh + h) * Dd * Ss;

  // 4 async b128 copies per thread per key block (2 for K, 2 for V).
  auto issueKV = [&](int t0, int p) {
#pragma unroll
    for (int i = 0; i < 2; ++i) {
      int idx = tid + i * 128, tr = idx >> 2, c = idx & 3;
      async_ld_b128(&Ks[p][tr][c * 8], Kb + (size_t)(t0 + tr) * Dd + c * 8);
    }
#pragma unroll
    for (int i = 0; i < 2; ++i) {
      int idx = tid + i * 128, d = idx >> 2, c = idx & 3;
      async_ld_b128(&Vt[p][d][c * 8], Vb + (size_t)d * Ss + t0 + c * 8);
    }
  };

  // Q fragments live in registers across the whole key loop.
  AF qa[2];
  {
    const _Float16* qrow = Qb + (size_t)(wrow0 + ln) * Dd;
#pragma unroll
    for (int kk = 0; kk < 2; ++kk) {
      qa[kk].u[0] = *(const uint4*)(qrow + kk * 32 + kbase);
      qa[kk].u[1] = *(const uint4*)(qrow + kk * 32 + 16 + kbase);
    }
  }

  v8f o[4] = {};
  float mrun[8], lrun[8];
#pragma unroll
  for (int r = 0; r < 8; ++r) { mrun[r] = -3.4e38f; lrun[r] = 0.f; }

  const int tend = m0 + 64;
  issueKV(0, 0);                               // prime the pipeline

  for (int t0 = 0; t0 < tend; t0 += 32) {
    const int p = (t0 >> 5) & 1;
    const bool nxt = (t0 + 32) < tend;
    if (nxt) issueKV(t0 + 32, p ^ 1);          // overlap next block with compute
    if (nxt) wait_async4(); else wait_async0();// current block's 4 copies done
    __syncthreads();

    if (t0 <= wrow_max) {   // scalar guard: EXEC all-ones inside
      float scf[2][8];
#pragma unroll
      for (int st = 0; st < 2; ++st) {
        const int tbase = t0 + st * 16;
        if (tbase > wrow_max) {   // scalar guard: subtile fully masked
#pragma unroll
          for (int r = 0; r < 8; ++r) scf[st][r] = -3.4e38f;
        } else {
          AF kb;
          v8f a = {};
          const _Float16* krow = &Ks[p][st * 16 + ln][0];
#pragma unroll
          for (int kk = 0; kk < 2; ++kk) {
            kb.u[0] = *(const uint4*)(krow + kk * 32 + kbase);
            kb.u[1] = *(const uint4*)(krow + kk * 32 + 16 + kbase);
            a = wmma16(qa[kk], kb, a);           // S += Q_chunk * K_chunk^T
          }
#pragma unroll
          for (int r = 0; r < 8; ++r) {
            const int rowg = wrow0 + r + 8 * half;
            const int colg = tbase + ln;
            const float s = a[r] * 0.125f;        // 1/sqrt(64)
            scf[st][r] = (colg > rowg) ? -3.4e38f : s;   // causal mask
          }
        }
      }
      // Online softmax: each matrix row spans the 16 lanes of a half-wave.
#pragma unroll
      for (int r = 0; r < 8; ++r) {
        float v0 = scf[0][r], v1 = scf[1][r];
        float mx = fmaxf(v0, v1);
#pragma unroll
        for (int off = 1; off < 16; off <<= 1)
          mx = fmaxf(mx, __shfl_xor(mx, off, 32));
        const float mnew = fmaxf(mrun[r], mx);
        const float p0 = __expf(v0 - mnew);
        const float p1 = __expf(v1 - mnew);
        float rs = p0 + p1;
#pragma unroll
        for (int off = 1; off < 16; off <<= 1)
          rs += __shfl_xor(rs, off, 32);
        const float alpha = __expf(mrun[r] - mnew);
        lrun[r] = lrun[r] * alpha + rs;
        mrun[r] = mnew;
#pragma unroll
        for (int dn = 0; dn < 4; ++dn) o[dn][r] *= alpha;
        const int rl = r + 8 * half;              // C-layout -> LDS row-major
        Ps[wave][rl][ln]      = (_Float16)p0;
        Ps[wave][rl][16 + ln] = (_Float16)p1;
      }
      // O += P(16x32) * V(32x64): 4 WMMAs over the d tiles.
      AF pa;
      pa.u[0] = *(const uint4*)&Ps[wave][ln][kbase];
      pa.u[1] = *(const uint4*)&Ps[wave][ln][16 + kbase];
#pragma unroll
      for (int dn = 0; dn < 4; ++dn) {
        AF vb;
        const _Float16* vrow = &Vt[p][dn * 16 + ln][0];
        vb.u[0] = *(const uint4*)(vrow + kbase);
        vb.u[1] = *(const uint4*)(vrow + 16 + kbase);
        o[dn] = wmma16(pa, vb, o[dn]);
      }
    }
    __syncthreads();   // protect buf[p] before iteration i+1 overwrites it
  }

  // Normalize and emit the concat row [B*S][H*D] in f16 for the final GEMM.
#pragma unroll
  for (int r = 0; r < 8; ++r) {
    const float inv = 1.f / lrun[r];
    const int s = wrow0 + r + 8 * half;
    _Float16* dst = attn + ((size_t)b * Ss + s) * (Hh * Dd) + h * Dd;
#pragma unroll
    for (int dn = 0; dn < 4; ++dn)
      dst[dn * 16 + ln] = (_Float16)(o[dn][r] * inv);
  }
}

// ---------------------------------------------------------------------------
// Kernel 3: output projection + ReLU.  out = relu(concat[4096x1024]*Wp + bp)
// Tile 128x128, wave tile 32x64.  A tile is already f16 -> async global->LDS
// copies; B register-staged f32->f16.
// ---------------------------------------------------------------------------
__global__ __launch_bounds__(256, 2) void k_proj(
    const _Float16* __restrict__ attn, const float* __restrict__ Wp,
    const float* __restrict__ bp, float* __restrict__ out)
{
  __shared__ __align__(16) _Float16 As[2][128][32];
  __shared__ __align__(16) _Float16 Bs[2][128][32];

  const int bm0 = blockIdx.x * 128;
  const int n0  = blockIdx.y * 128;
  const int tid = threadIdx.x;
  const int wave = tid >> 5, lane = tid & 31;
  const int half = lane >> 4, ln = lane & 15;
  const int wm = wave & 3, wn = wave >> 2;
  const int kbase = half * 8;
  const int HD = Hh * Dd;

  float4 fb[4];
  auto asyncA = [&](int k0, int p) {           // 2 async b128 per thread
#pragma unroll
    for (int i = 0; i < 2; ++i) {
      int idx = tid + i * 256, row = idx >> 2, c = idx & 3;
      async_ld_b128(&As[p][row][c * 8],
                    attn + (size_t)(bm0 + row) * HD + k0 + c * 8);
    }
  };
  auto loadB = [&](int k0) {
#pragma unroll
    for (int i = 0; i < 4; ++i) {              // 32(k) x 128(n) f32
      int idx = tid + i * 256, e = idx >> 5, c = idx & 31;
      fb[i] = *(const float4*)(Wp + (size_t)(k0 + e) * Ee + n0 + c * 4);
    }
  };
  auto storeB = [&](int p) {
#pragma unroll
    for (int i = 0; i < 4; ++i) {              // Bs[n][k] = Wp[k][n]
      int idx = tid + i * 256, e = idx >> 5, c = idx & 31;
      Bs[p][c * 4 + 0][e] = (_Float16)fb[i].x;
      Bs[p][c * 4 + 1][e] = (_Float16)fb[i].y;
      Bs[p][c * 4 + 2][e] = (_Float16)fb[i].z;
      Bs[p][c * 4 + 3][e] = (_Float16)fb[i].w;
    }
  };

  asyncA(0, 0);
  loadB(0);
  storeB(0);
  wait_async0();
  __syncthreads();

  v8f acc[2][4] = {};
  for (int k0 = 0; k0 < HD; k0 += 32) {
    const int p = (k0 >> 5) & 1;
    const bool nxt = (k0 + 32) < HD;
    if (nxt) { asyncA(k0 + 32, p ^ 1); loadB(k0 + 32); }

    AF a[2], bf[4];
#pragma unroll
    for (int t = 0; t < 2; ++t) {
      const int row = wm * 32 + t * 16 + ln;
      a[t].u[0] = *(const uint4*)&As[p][row][kbase];
      a[t].u[1] = *(const uint4*)&As[p][row][16 + kbase];
    }
#pragma unroll
    for (int t = 0; t < 4; ++t) {
      const int col = wn * 64 + t * 16 + ln;
      bf[t].u[0] = *(const uint4*)&Bs[p][col][kbase];
      bf[t].u[1] = *(const uint4*)&Bs[p][col][16 + kbase];
    }
#pragma unroll
    for (int tm = 0; tm < 2; ++tm)
#pragma unroll
      for (int tn = 0; tn < 4; ++tn)
        acc[tm][tn] = wmma16(a[tm], bf[tn], acc[tm][tn]);

    if (nxt) {
      storeB(p ^ 1);
      wait_async0();                           // next A tile landed during WMMA
      __syncthreads();
    }
  }

#pragma unroll
  for (int tm = 0; tm < 2; ++tm)
#pragma unroll
    for (int tn = 0; tn < 4; ++tn) {
      const int n = n0 + wn * 64 + tn * 16 + ln;
      const float bias = bp[n];
#pragma unroll
      for (int r = 0; r < 8; ++r) {
        const int m = bm0 + wm * 32 + tm * 16 + r + 8 * half;
        const float v = acc[tm][tn][r] + bias;
        out[(size_t)m * Ee + n] = v > 0.f ? v : 0.f;
      }
    }
}

// ---------------------------------------------------------------------------
extern "C" void kernel_launch(void* const* d_in, const int* in_sizes, int n_in,
                              void* d_out, int out_size, void* d_ws,
                              size_t ws_size, hipStream_t stream) {
  (void)in_sizes; (void)n_in; (void)out_size; (void)ws_size;
  const float* x  = (const float*)d_in[0];
  const float* Wq = (const float*)d_in[1];
  const float* Wk = (const float*)d_in[2];
  const float* Wv = (const float*)d_in[3];
  const float* bq = (const float*)d_in[4];
  const float* bk = (const float*)d_in[5];
  const float* bv = (const float*)d_in[6];
  const float* Wp = (const float*)d_in[7];
  const float* bp = (const float*)d_in[8];
  float* out = (float*)d_out;

  const size_t nQKV = (size_t)Bb * Hh * Ss * Dd;   // 4,194,304 halves each
  _Float16* Qh   = (_Float16*)d_ws;                // 8 MB
  _Float16* Kh   = Qh + nQKV;                      // 8 MB
  _Float16* Vh   = Kh + nQKV;                      // 8 MB (transposed [B,H,D,S])
  _Float16* attn = Vh + nQKV;                      // 8 MB concat rows

  k_qkv <<<dim3((Bb * Ss) / 128, Hh / 2, 3), 256, 0, stream>>>(
      x, Wq, Wk, Wv, bq, bk, bv, Qh, Kh, Vh);
  k_flash<<<dim3(Ss / 64, Hh, Bb), 128, 0, stream>>>(Qh, Kh, Vh, attn);
  k_proj <<<dim3((Bb * Ss) / 128, Ee / 128), 256, 0, stream>>>(attn, Wp, bp, out);
}
